// ETypePromptModel_80135499809109
// MI455X (gfx1250) — compile-verified, matrix operand order checked by reading, not run on hardware
//
#include <hip/hip_runtime.h>

// Problem constants (from reference): B=256, S=512, D=768, L=8, C=4096
#define B_  256
#define S_  512
#define D_  768
#define L_  8
#define C_  4096
#define K_  (L_ * D_)          // 6144  (flattened l,d contraction dim)

// GEMM tiling
#define BM      128            // M tile (batch rows) per workgroup
#define BN      64             // N tile (classes) per workgroup
#define BK      64             // K step per stage (2 x wmma depth 32)
#define KSTEPS  (K_ / BK)      // 96
#define LDS_STRIDE_U 36        // 64 bf16 = 32 uints, +4 pad = 144 B (16B-aligned, bank-spread)

typedef __attribute__((ext_vector_type(16))) __bf16 v16bf;
typedef __attribute__((ext_vector_type(8)))  float  v8f;

// f32 pair -> packed bf16x2 (round-to-nearest, ties-up) in 3 VALU ops:
// two v_add_u32 + one v_perm_b32 picking the high halves.
__device__ __forceinline__ unsigned int pack_bf16x2(float lo, float hi) {
    unsigned int a = __float_as_uint(hi) + 0x8000u;   // rounded hi
    unsigned int b = __float_as_uint(lo) + 0x8000u;   // rounded lo
    // v_perm_b32: {S0,S1} bytes, S1 = bytes 0-3. Result = {a[31:16], b[31:16]}.
    return __builtin_amdgcn_perm(a, b, 0x07060302u);
}

// ---------------------------------------------------------------------------
// Kernel 1: gather mlm positions and convert f32 -> packed bf16 rows.
// Hb layout: [B][K_] bf16 stored as [B][K_/2] uints (K contiguous per row).
// One block per (b,l); copies hidden_states[b, pos[b,l], :] (768 floats).
// ---------------------------------------------------------------------------
__global__ void __launch_bounds__(128)
gather_cvt_kernel(const float* __restrict__ hidden,
                  const int* __restrict__ pos,
                  unsigned int* __restrict__ Hb) {
    const int bl = blockIdx.x;              // 0 .. B*L-1
    const int b  = bl / L_;
    const int l  = bl % L_;
    const int p  = pos[bl];                 // 0 .. S-1
    const float2* __restrict__ src =
        (const float2*)(hidden + ((size_t)b * S_ + p) * D_);
    unsigned int* __restrict__ dst = Hb + (size_t)b * (K_ / 2) + l * (D_ / 2);
    for (int j = threadIdx.x; j < D_ / 2; j += 128) {
        float2 f = src[j];
        dst[j] = pack_bf16x2(f.x, f.y);
    }
}

// ---------------------------------------------------------------------------
// Kernel 2: logits[b,c] = sum_k Hb[b,k] * E[c,k]  via v_wmma_f32_16x16x32_bf16
// Grid: (C/BN, B/BM) = (64, 2). Block: 256 threads = 8 wave32.
// Wave grid 4(M) x 2(N); each wave owns a 32x32 micro-tile -> 8 wmma / K-step.
// Double-buffered LDS; E converted f32->bf16 on the fly during staging.
// ---------------------------------------------------------------------------
__global__ void __launch_bounds__(256)
gemm_bf16_wmma_kernel(const unsigned int* __restrict__ Hb,
                      const float* __restrict__ E,
                      float* __restrict__ out) {
    __shared__ __align__(16) unsigned int As[2][BM * LDS_STRIDE_U]; // 36 KB
    __shared__ __align__(16) unsigned int Bs[2][BN * LDS_STRIDE_U]; // 18 KB

    const int tid   = threadIdx.x;
    const int Nbase = blockIdx.x * BN;
    const int Mbase = blockIdx.y * BM;

    const int wave = tid >> 5;
    const int lane = tid & 31;
    const int wm   = wave >> 1;      // 0..3 : 32-row band
    const int wn   = wave & 1;       // 0..1 : 32-col band
    const int l16  = lane & 15;      // row/col within 16x16 tile
    const int hi   = lane >> 4;      // K-half selector (ISA bf16 A/B layout)

    // Stage one K-step (global -> LDS), converting E to bf16 in flight.
    auto stage = [&](int buf, int k0) {
        // A tile: 128 rows x 64 bf16 = 1024 uint4 (bf16 already packed).
        const uint4* __restrict__ Ag = (const uint4*)Hb;   // K_/8 uint4 per row
#pragma unroll
        for (int i = 0; i < 4; ++i) {
            int e   = tid + i * 256;
            int row = e >> 3, c4 = e & 7;                  // 8 uint4 per row
            uint4 v = Ag[(size_t)(Mbase + row) * (K_ / 8) + (k0 >> 3) + c4];
            *(uint4*)&As[buf][row * LDS_STRIDE_U + c4 * 4] = v;
        }
        // B tile: 64 classes x 64 f32 = 1024 float4 -> bf16 pairs.
#pragma unroll
        for (int i = 0; i < 4; ++i) {
            int e  = tid + i * 256;
            int cr = e >> 4, f4 = e & 15;                  // 16 float4 per row
            const float4* __restrict__ gp =
                (const float4*)(E + (size_t)(Nbase + cr) * K_ + k0) + f4;
            if (k0 + BK < K_)  // prefetch next K-step of E (global_prefetch_b8)
                __builtin_prefetch((const void*)(gp + (BK / 4)), 0, 1);
            float4 v = *gp;
            uint2 u;
            u.x = pack_bf16x2(v.x, v.y);
            u.y = pack_bf16x2(v.z, v.w);
            *(uint2*)&Bs[buf][cr * LDS_STRIDE_U + f4 * 2] = u;
        }
    };

    v8f acc[2][2] = {};   // four 16x16 f32 accumulators (8 VGPRs each)

    stage(0, 0);
    __syncthreads();

    for (int ks = 0; ks < KSTEPS; ++ks) {
        const int cur = ks & 1;
        if (ks + 1 < KSTEPS) stage(cur ^ 1, (ks + 1) * BK);

        const uint4* __restrict__ A4 = (const uint4*)&As[cur][0]; // 9 uint4/row
        const uint4* __restrict__ B4 = (const uint4*)&Bs[cur][0];

        union Frag { uint4 u[2]; v16bf v; };
        // Two wmma K-chunks of 32 within this 64-wide stage.
#pragma unroll
        for (int kc = 0; kc < 2; ++kc) {
            Frag aF[2], bF[2];
            // ISA bf16 fragment layout: lane<16 -> K 0..7 (V0-3) + 16..23
            // (V4-7); lane>=16 -> K 8..15 + 24..31. Two b128 LDS loads each.
#pragma unroll
            for (int tm = 0; tm < 2; ++tm) {
                int row = wm * 32 + tm * 16 + l16;
                aF[tm].u[0] = A4[row * 9 + kc * 4 + hi];
                aF[tm].u[1] = A4[row * 9 + kc * 4 + hi + 2];
            }
#pragma unroll
            for (int tn = 0; tn < 2; ++tn) {
                int col = wn * 32 + tn * 16 + l16;
                bF[tn].u[0] = B4[col * 9 + kc * 4 + hi];
                bF[tn].u[1] = B4[col * 9 + kc * 4 + hi + 2];
            }
#pragma unroll
            for (int tm = 0; tm < 2; ++tm)
#pragma unroll
                for (int tn = 0; tn < 2; ++tn)
                    acc[tm][tn] = __builtin_amdgcn_wmma_f32_16x16x32_bf16(
                        false, aF[tm].v, false, bF[tn].v,
                        (short)0, acc[tm][tn], false, false);
        }

        __syncthreads();
    }

    // Epilogue: f32 C/D layout -> VGPR i holds M = i (lanes 0-15) / 8+i (16-31).
#pragma unroll
    for (int tm = 0; tm < 2; ++tm) {
#pragma unroll
        for (int tn = 0; tn < 2; ++tn) {
            const int n  = Nbase + wn * 32 + tn * 16 + l16;
            const int mb = Mbase + wm * 32 + tm * 16 + hi * 8;
#pragma unroll
            for (int i = 0; i < 8; ++i)
                out[(size_t)(mb + i) * C_ + n] = acc[tm][tn][i];
        }
    }
}

// ---------------------------------------------------------------------------
// Launch: inputs are (hidden_states f32 [B,S,D], mlm_positions i32 [B,L],
// label2embed f32 [C,L,D]); output f32 [B,C]. Workspace holds bf16 H (3 MiB).
// ---------------------------------------------------------------------------
extern "C" void kernel_launch(void* const* d_in, const int* in_sizes, int n_in,
                              void* d_out, int out_size, void* d_ws, size_t ws_size,
                              hipStream_t stream) {
    (void)in_sizes; (void)n_in; (void)out_size; (void)ws_size;
    const float* hidden = (const float*)d_in[0];
    const int*   pos    = (const int*)d_in[1];
    const float* E      = (const float*)d_in[2];
    float*       out    = (float*)d_out;
    unsigned int* Hb    = (unsigned int*)d_ws;   // B * K/2 uints = 3.0 MiB

    gather_cvt_kernel<<<B_ * L_, 128, 0, stream>>>(hidden, pos, Hb);

    dim3 grid(C_ / BN, B_ / BM);   // (64, 2)
    gemm_bf16_wmma_kernel<<<grid, 256, 0, stream>>>(Hb, E, out);
}
